// DynamicConv2d_13675175870500
// MI455X (gfx1250) — matrix operand order repaired and checked by reading, not compiled, not run on hardware
//
#include <hip/hip_runtime.h>
#include <hip/hip_bf16.h>

typedef __attribute__((ext_vector_type(16))) __bf16 v16bf;
typedef __attribute__((ext_vector_type(8)))  float  v8f;

#define B_   32
#define CIN  128
#define COUT 128
#define HH   64
#define WW_  64
#define EE   4
#define HWD  (HH * WW_)
#define HP   66            // padded height
#define WP   66            // padded width

// workspace layout (bytes)
static const size_t OFF_POOLED = 0;                                    // B*CIN f32   (16384)
static const size_t OFF_ATTN   = OFF_POOLED + (size_t)B_ * CIN * 4;    // B*E  f32    (512)
static const size_t OFF_AGGB   = OFF_ATTN   + (size_t)B_ * EE  * 4;    // B*COUT f32  (16384)
static const size_t OFF_XB     = 33280;                                // B*HP*WP*CIN bf16 (35.7MB)
static const size_t XB_BYTES   = (size_t)B_ * HP * WP * CIN * 2;
static const size_t OFF_AGGW   = OFF_XB + XB_BYTES;                    // B*9*COUT*CIN bf16 (9.4MB)

// ---------------- router path ----------------

__global__ __launch_bounds__(256) void pool_kernel(const float* __restrict__ x,
                                                   float* __restrict__ pooled) {
  __shared__ float red[256];
  const int plane = blockIdx.x;                 // b*CIN + ci
  const float* p = x + (size_t)plane * HWD;
  float s = 0.f;
  for (int i = threadIdx.x; i < HWD; i += 256) s += p[i];
  red[threadIdx.x] = s;
  __syncthreads();
  for (int off = 128; off > 0; off >>= 1) {
    if ((int)threadIdx.x < off) red[threadIdx.x] += red[threadIdx.x + off];
    __syncthreads();
  }
  if (threadIdx.x == 0) pooled[plane] = red[0] * (1.0f / HWD);
}

__global__ void router_kernel(const float* __restrict__ pooled,
                              const float* __restrict__ rw,
                              const float* __restrict__ rb,
                              float* __restrict__ attn) {
  const int b = threadIdx.x;
  if (b >= B_) return;
  float lg[EE];
  for (int e = 0; e < EE; ++e) {
    float s = rb[e];
    for (int c = 0; c < CIN; ++c) s += pooled[b * CIN + c] * rw[e * CIN + c];
    lg[e] = s;
  }
  float m = lg[0];
  for (int e = 1; e < EE; ++e) m = fmaxf(m, lg[e]);
  float den = 0.f;
  for (int e = 0; e < EE; ++e) { lg[e] = __expf(lg[e] - m); den += lg[e]; }
  const float inv = 1.0f / den;
  for (int e = 0; e < EE; ++e) attn[b * EE + e] = lg[e] * inv;
}

__global__ void aggb_kernel(const float* __restrict__ attn,
                            const float* __restrict__ bias,
                            float* __restrict__ aggb) {
  const int idx = blockIdx.x * 256 + threadIdx.x;   // b*COUT + co
  const int b = idx / COUT, co = idx % COUT;
  float s = 0.f;
  #pragma unroll
  for (int e = 0; e < EE; ++e) s += attn[b * EE + e] * bias[e * COUT + co];
  aggb[idx] = s;
}

// zero the padded NHWC buffer (vectorized)
__global__ __launch_bounds__(256) void zero_xb_kernel(uint4* __restrict__ p, size_t n16) {
  for (size_t i = (size_t)blockIdx.x * 256 + threadIdx.x; i < n16; i += (size_t)4096 * 256)
    p[i] = make_uint4(0, 0, 0, 0);
}

// NCHW f32 -> padded NHWC bf16 via LDS tile transpose.
// block = (b, ci-group of 32, image row h). Reads coalesced along w, writes
// coalesced along ci.
__global__ __launch_bounds__(256) void convert_x_kernel(const float* __restrict__ x,
                                                        __bf16* __restrict__ xb) {
  const int blk = blockIdx.x;
  const int h  = blk & 63;
  const int cg = (blk >> 6) & 3;     // ci group (32 each)
  const int b  = blk >> 8;
  __shared__ float tile[32][65];     // +1 pad: conflict-free transposed reads
  const int t = threadIdx.x;
  {
    const int col = t & 63;
    const int cr  = t >> 6;          // 0..3
    #pragma unroll
    for (int j = 0; j < 8; ++j) {
      const int ci = cr + j * 4;     // 0..31
      tile[ci][col] = x[(size_t)(b * CIN + cg * 32 + ci) * HWD + h * WW_ + col];
    }
  }
  __syncthreads();
  {
    const int ci = t & 31;
    const int wr = t >> 5;           // 0..7
    #pragma unroll
    for (int j = 0; j < 8; ++j) {
      const int w = wr + j * 8;      // 0..63
      xb[(((size_t)b * HP + (h + 1)) * WP + (w + 1)) * CIN + cg * 32 + ci] =
          (__bf16)tile[ci][w];
    }
  }
}

// expert mix -> bf16, layout [b][tap][co][ci]
__global__ __launch_bounds__(256) void aggw_kernel(const float* __restrict__ attn,
                                                   const float* __restrict__ w,
                                                   __bf16* __restrict__ aw) {
  const size_t idx = (size_t)blockIdx.x * 256 + threadIdx.x;
  const int ci = (int)(idx % CIN);
  size_t t = idx / CIN;
  const int co  = (int)(t % COUT); t /= COUT;
  const int tap = (int)(t % 9);
  const int b   = (int)(t / 9);
  float s = 0.f;
  #pragma unroll
  for (int e = 0; e < EE; ++e)
    s += attn[b * EE + e] * w[(((size_t)e * COUT + co) * CIN + ci) * 9 + tap];
  aw[idx] = (__bf16)s;
}

// ---------------- WMMA conv: 9 shifted GEMMs, branch-free ----------------
// block = (b, output row h); 8 waves: 4 M-quadrants x 2 N-halves;
// wave computes a 32(co) x 32(w) tile as 2x2 wmma 16x16 f32 accumulators.
// Padded NHWC x => every fragment load is an unconditional global_load_b128.

__global__ __launch_bounds__(256) void conv_kernel(const __bf16* __restrict__ xb,
                                                   const __bf16* __restrict__ aw,
                                                   const float* __restrict__ aggb,
                                                   float* __restrict__ out) {
  const int b    = blockIdx.x >> 6;
  const int h    = blockIdx.x & 63;
  const int tid  = threadIdx.x;
  const int lane = tid & 31;
  const int wv   = tid >> 5;    // 0..7
  const int mw   = wv & 3;      // co quadrant (32 rows each)
  const int nw   = wv >> 2;     // n half (32 cols each)
  const int half = lane >> 4;
  const int lm   = lane & 15;

  v8f acc[2][2] = {};
  union Frag { uint4 u[2]; v16bf v; };

  const __bf16* xbase = xb + ((size_t)b * HP + h) * (WP * CIN); // padded row h+dh = +dh*WP*CIN
  const __bf16* awb   = aw + (size_t)b * 9 * COUT * CIN;

  #pragma unroll 1
  for (int tap = 0; tap < 9; ++tap) {
    const int dh = tap / 3, dw = tap % 3;           // padded offsets (0..2)
    const __bf16* awp  = awb + (size_t)tap * COUT * CIN;
    const __bf16* xrow = xbase + (size_t)dh * (WP * CIN);

    #pragma unroll
    for (int c = 0; c < CIN; c += 32) {
      Frag A[2], Bf[2];
      // A: agg_w[co][ci]; lane holds K = half*8+0..7 , 16+half*8+0..7
      #pragma unroll
      for (int mi = 0; mi < 2; ++mi) {
        const int co = mw * 32 + mi * 16 + lm;
        const __bf16* p = awp + (size_t)co * CIN + c + half * 8;
        A[mi].u[0] = *reinterpret_cast<const uint4*>(p);
        A[mi].u[1] = *reinterpret_cast<const uint4*>(p + 16);
      }
      // B: x[ci][w]; lane holds K = half*8+0..7 , 16+half*8+0..7 at padded column n+dw
      #pragma unroll
      for (int ni = 0; ni < 2; ++ni) {
        const int wp = nw * 32 + ni * 16 + lm + dw;  // 0..65, always in-bounds
        const __bf16* p = xrow + (size_t)wp * CIN + c + half * 8;
        Bf[ni].u[0] = *reinterpret_cast<const uint4*>(p);
        Bf[ni].u[1] = *reinterpret_cast<const uint4*>(p + 16);
      }
      #pragma unroll
      for (int mi = 0; mi < 2; ++mi)
        #pragma unroll
        for (int ni = 0; ni < 2; ++ni)
          acc[mi][ni] = __builtin_amdgcn_wmma_f32_16x16x32_bf16(
              false, A[mi].v, false, Bf[ni].v, (short)0, acc[mi][ni], false, false);
    }
  }

  // epilogue: + agg_b, store f32 NCHW (lanes 0..15 store 64B-contiguous runs)
  const float* bb = aggb + b * COUT;
  #pragma unroll
  for (int mi = 0; mi < 2; ++mi) {
    #pragma unroll
    for (int r = 0; r < 8; ++r) {
      const int co = mw * 32 + mi * 16 + half * 8 + r;
      const float bv = bb[co];
      #pragma unroll
      for (int ni = 0; ni < 2; ++ni) {
        const int col = nw * 32 + ni * 16 + lm;
        out[(size_t)(b * COUT + co) * HWD + h * WW_ + col] = acc[mi][ni][r] + bv;
      }
    }
  }
}

// ---------------- launcher ----------------

extern "C" void kernel_launch(void* const* d_in, const int* in_sizes, int n_in,
                              void* d_out, int out_size, void* d_ws, size_t ws_size,
                              hipStream_t stream) {
  const float* x      = (const float*)d_in[0];
  const float* weight = (const float*)d_in[1];
  const float* bias   = (const float*)d_in[2];
  const float* rw     = (const float*)d_in[3];
  const float* rb     = (const float*)d_in[4];
  float* out = (float*)d_out;

  char* ws = (char*)d_ws;
  float*  pooled = (float*)(ws + OFF_POOLED);
  float*  attn   = (float*)(ws + OFF_ATTN);
  float*  aggb   = (float*)(ws + OFF_AGGB);
  __bf16* xb     = (__bf16*)(ws + OFF_XB);
  __bf16* aw     = (__bf16*)(ws + OFF_AGGW);

  pool_kernel<<<B_ * CIN, 256, 0, stream>>>(x, pooled);
  router_kernel<<<1, 32, 0, stream>>>(pooled, rw, rb, attn);
  aggb_kernel<<<(B_ * COUT) / 256, 256, 0, stream>>>(attn, bias, aggb);
  zero_xb_kernel<<<4096, 256, 0, stream>>>((uint4*)xb, XB_BYTES / 16);
  convert_x_kernel<<<B_ * 4 * HH, 256, 0, stream>>>(x, xb);
  aggw_kernel<<<(B_ * 9 * COUT * CIN) / 256, 256, 0, stream>>>(attn, weight, aw);
  conv_kernel<<<B_ * HH, 256, 0, stream>>>(xb, aw, aggb, out);
}
